// GCN_22462678958497
// MI455X (gfx1250) — matrix operand order repaired and checked by reading, not compiled, not run on hardware
//
#include <hip/hip_runtime.h>
#include <hip/hip_bf16.h>

typedef __attribute__((ext_vector_type(16))) _Float16 v16h;
typedef __attribute__((ext_vector_type(8)))  float    v8f;

#define HDIM 20
#define FDIM 128

// ---------------- utility kernels ----------------

__global__ void k_zero_f32(float* __restrict__ p, int n) {
    int i = blockIdx.x * blockDim.x + threadIdx.x;
    if (i < n) p[i] = 0.0f;
}

// deg[dst] += w  (edge part of degree; self-loop +1 folded into finalize)
__global__ void k_deg_edges(const int* __restrict__ dst, const float* __restrict__ ew,
                            float* __restrict__ deg, int E) {
    int e = blockIdx.x * blockDim.x + threadIdx.x;
    if (e < E) unsafeAtomicAdd(&deg[dst[e]], ew[e]);
}

// in-place: dinv = rsqrt(deg + 1)   (deg+1 > 0 always -> matches where(deg>0, rsqrt, 0))
__global__ void k_finalize_dinv(float* __restrict__ d, int n) {
    int i = blockIdx.x * blockDim.x + threadIdx.x;
    if (i < n) d[i] = rsqrtf(d[i] + 1.0f);
}

__global__ void k_edge_norm(const int* __restrict__ src, const int* __restrict__ dst,
                            const float* __restrict__ ew, const float* __restrict__ dinv,
                            float* __restrict__ nrm, int E) {
    int e = blockIdx.x * blockDim.x + threadIdx.x;
    if (e < E) nrm[e] = dinv[src[e]] * ew[e] * dinv[dst[e]];
}

// ---------------- B-fragment pre-pack ----------------
// Pack W[K,HDIM] (f32, K zero-padded to nkk*32) into f16 fragments laid out so each
// lane's 16 halves are contiguous (32B):  Bp[((kk*2+nt)*32 + lane)*16 + i].
// Dense 32x16 f16 B layout: lanes 0-15 col=lane hold K=0..7,16..23,
// lanes 16-31 col=lane-16 hold K=8..15,24..31 (matches A-frag order).
__global__ void k_pack_b(const float* __restrict__ W, _Float16* __restrict__ Bp,
                         int K, int nkk) {
    int t = blockIdx.x * blockDim.x + threadIdx.x;
    if (t >= nkk * 2 * 32 * 16) return;
    int i    = t & 15;
    int lane = (t >> 4) & 31;
    int nt   = (t >> 9) & 1;
    int kk   = t >> 10;
    int col  = nt * 16 + (lane & 15);
    int k    = kk * 32 + (lane >> 4) * 8 + (i & 7) + (i >> 3) * 16;
    Bp[t] = (_Float16)((k < K && col < HDIM) ? W[k * HDIM + col] : 0.0f);
}

// ---------------- WMMA GEMMs ----------------
// A-frag (16-bit A 16x32): lane -> row = lane&15, base = (lane>=16)?8:0;
//   elements 0..7  <- K = base+0 .. base+7   (contiguous)
//   elements 8..15 <- K = base+16 .. base+23 (contiguous)
// D (f32 16x16): VGPR r, lane -> row = (lane>>4)*8 + r, col = lane&15.
// Epilogue writes P (raw projection, gathered by edge kernel) and
// AGG = dinv^2 * P + bias (self-loop + bias, initializes the scatter target).

__device__ __forceinline__ void a_from_f4(v16h& a, float4 p0, float4 p1,
                                          float4 p2, float4 p3) {
    a[0]=(_Float16)p0.x;  a[1]=(_Float16)p0.y;  a[2]=(_Float16)p0.z;  a[3]=(_Float16)p0.w;
    a[4]=(_Float16)p1.x;  a[5]=(_Float16)p1.y;  a[6]=(_Float16)p1.z;  a[7]=(_Float16)p1.w;
    a[8]=(_Float16)p2.x;  a[9]=(_Float16)p2.y;  a[10]=(_Float16)p2.z; a[11]=(_Float16)p2.w;
    a[12]=(_Float16)p3.x; a[13]=(_Float16)p3.y; a[14]=(_Float16)p3.z; a[15]=(_Float16)p3.w;
}

__device__ __forceinline__ void store_epilogue(v8f c, int col, int rbase, int nrows,
                                               bool full, const float* __restrict__ di2,
                                               const float* __restrict__ bias,
                                               float* __restrict__ P,
                                               float* __restrict__ AGG) {
    if (col >= HDIM) return;
    float bv = bias[col];
    if (full) {  // wave-uniform fast path: straight-line stores, no exec churn
#pragma unroll
        for (int r = 0; r < 8; ++r) {
            float pv = c[r];
            size_t o = (size_t)(rbase + r) * HDIM + col;
            P[o]   = pv;
            AGG[o] = di2[r] * pv + bv;
        }
    } else {
#pragma unroll
        for (int r = 0; r < 8; ++r) {
            int mr = rbase + r;
            if (mr < nrows) {
                float pv = c[r];
                P[mr * HDIM + col]   = pv;
                AGG[mr * HDIM + col] = di2[r] * pv + bv;
            }
        }
    }
}

// Layer 1: X[N,128] @ W1 -> P, AGG
__global__ void k_gemm_f128(const float* __restrict__ X, const _Float16* __restrict__ Bp,
                            const float* __restrict__ dinv, const float* __restrict__ bias,
                            float* __restrict__ P, float* __restrict__ AGG, int nrows) {
    int wave = (blockIdx.x * blockDim.x + threadIdx.x) >> 5;
    int lane = threadIdx.x & 31;
    int tileM = wave;
    if (tileM * 16 >= nrows) return;
    bool full = (tileM * 16 + 16) <= nrows;               // uniform across wave
    int row  = min(tileM * 16 + (lane & 15), nrows - 1);  // clamp: extra rows never stored
    int base = ((lane >> 4) & 1) * 8;
    const float* xr = X + (size_t)row * FDIM;

    v16h a[4];
#pragma unroll
    for (int kk = 0; kk < 4; ++kk) {
        const float* xp = xr + kk * 32 + base;
        float4 p0 = *(const float4*)(xp);
        float4 p1 = *(const float4*)(xp + 4);
        float4 p2 = *(const float4*)(xp + 16);
        float4 p3 = *(const float4*)(xp + 20);
        a_from_f4(a[kk], p0, p1, p2, p3);
    }

    int rbase = tileM * 16 + ((lane >> 4) << 3);
    float di2[8];
#pragma unroll
    for (int r = 0; r < 8; ++r) {
        float di = dinv[min(rbase + r, nrows - 1)];
        di2[r] = di * di;
    }

#pragma unroll
    for (int nt = 0; nt < 2; ++nt) {
        v8f c = {};
#pragma unroll
        for (int kk = 0; kk < 4; ++kk) {
            const v16h b = *(const v16h*)(Bp + (((kk << 1) | nt) * 32 + lane) * 16);
            c = __builtin_amdgcn_wmma_f32_16x16x32_f16(false, a[kk], false, b,
                                                       (short)0, c, false, false);
        }
        store_epilogue(c, nt * 16 + (lane & 15), rbase, nrows, full, di2, bias, P, AGG);
    }
}

// Layers 2/3: Hin[N,20] @ W -> P, AGG   (K padded 20->32 with zeros)
__global__ void k_gemm_f20(const float* __restrict__ Hin, const _Float16* __restrict__ Bp,
                           const float* __restrict__ dinv, const float* __restrict__ bias,
                           float* __restrict__ P, float* __restrict__ AGG, int nrows) {
    int wave = (blockIdx.x * blockDim.x + threadIdx.x) >> 5;
    int lane = threadIdx.x & 31;
    int tileM = wave;
    if (tileM * 16 >= nrows) return;
    bool full = (tileM * 16 + 16) <= nrows;
    int row  = min(tileM * 16 + (lane & 15), nrows - 1);
    int base = ((lane >> 4) & 1) * 8;
    const float* hr = Hin + (size_t)row * HDIM;

    // base=0: K 0..7 | 16..19 valid (20..23 pad); base=8: K 8..15 valid (24..31 pad)
    float4 zero4 = make_float4(0.f, 0.f, 0.f, 0.f);
    float4 p0 = *(const float4*)(hr + base);
    float4 p1 = *(const float4*)(hr + base + 4);
    float4 p2 = (base == 0) ? *(const float4*)(hr + 16) : zero4;
    v16h a;
    a_from_f4(a, p0, p1, p2, zero4);

    int rbase = tileM * 16 + ((lane >> 4) << 3);
    float di2[8];
#pragma unroll
    for (int r = 0; r < 8; ++r) {
        float di = dinv[min(rbase + r, nrows - 1)];
        di2[r] = di * di;
    }

#pragma unroll
    for (int nt = 0; nt < 2; ++nt) {
        const v16h b = *(const v16h*)(Bp + (nt * 32 + lane) * 16);
        v8f c = {};
        c = __builtin_amdgcn_wmma_f32_16x16x32_f16(false, a, false, b,
                                                   (short)0, c, false, false);
        store_epilogue(c, nt * 16 + (lane & 15), rbase, nrows, full, di2, bias, P, AGG);
    }
}

// ---------------- aggregation ----------------

// agg[dst] += norm * P[src]  (one thread per edge, float4 gather, native f32 atomics)
__global__ void k_edge_agg(const int* __restrict__ src, const int* __restrict__ dst,
                           const float* __restrict__ nrm, const float* __restrict__ P,
                           float* __restrict__ agg, int E) {
    int e = blockIdx.x * blockDim.x + threadIdx.x;
    if (e >= E) return;
    int s = src[e], d = dst[e];
    float nm = nrm[e];
    const float4* pr = (const float4*)(P + (size_t)s * HDIM);  // 80B rows stay 16B aligned
    float* ar = agg + (size_t)d * HDIM;
#pragma unroll
    for (int q = 0; q < 5; ++q) {
        float4 v = pr[q];
        unsafeAtomicAdd(ar + 4 * q + 0, nm * v.x);
        unsafeAtomicAdd(ar + 4 * q + 1, nm * v.y);
        unsafeAtomicAdd(ar + 4 * q + 2, nm * v.z);
        unsafeAtomicAdd(ar + 4 * q + 3, nm * v.w);
    }
}

// Hout = relu(agg / max(||agg||, eps))   (one thread per node, float4 row I/O)
__global__ void k_norm_relu(const float* __restrict__ agg, float* __restrict__ Hout, int n) {
    int i = blockIdx.x * blockDim.x + threadIdx.x;
    if (i >= n) return;
    const float4* ar = (const float4*)(agg + (size_t)i * HDIM);
    float4 v[5];
    float ss = 0.0f;
#pragma unroll
    for (int q = 0; q < 5; ++q) {
        v[q] = ar[q];
        ss += v[q].x * v[q].x + v[q].y * v[q].y + v[q].z * v[q].z + v[q].w * v[q].w;
    }
    float inv = 1.0f / fmaxf(sqrtf(ss), 1e-12f);
    float4* hr = (float4*)(Hout + (size_t)i * HDIM);
#pragma unroll
    for (int q = 0; q < 5; ++q) {
        float4 o;
        o.x = fmaxf(v[q].x * inv, 0.0f);
        o.y = fmaxf(v[q].y * inv, 0.0f);
        o.z = fmaxf(v[q].z * inv, 0.0f);
        o.w = fmaxf(v[q].w * inv, 0.0f);
        hr[q] = o;
    }
}

// ---------------- pooling + classifier ----------------

// int-bit atomicMax is order-preserving for non-negative post-ReLU values;
// 0-init reproduces reference's where(isfinite, ., 0) for empty groups.
__global__ void k_pool_max(const float* __restrict__ Hin, const int* __restrict__ batch,
                           float* __restrict__ pooled, int n) {
    int t = blockIdx.x * blockDim.x + threadIdx.x;
    if (t >= n * HDIM) return;
    int i = t / HDIM, f = t - i * HDIM;
    int g = batch[i];
    atomicMax((int*)&pooled[g * HDIM + f], __float_as_int(Hin[t]));
}

__global__ void k_linear(const float* __restrict__ pooled, const float* __restrict__ W,
                         const float* __restrict__ b, float* __restrict__ out, int G) {
    int t = blockIdx.x * blockDim.x + threadIdx.x;
    if (t >= G * 10) return;
    int g = t / 10, c = t - g * 10;
    float acc = b[c];
#pragma unroll
    for (int f = 0; f < HDIM; ++f) acc += pooled[g * HDIM + f] * W[f * 10 + c];
    out[t] = acc;
}

// ---------------- launch ----------------

extern "C" void kernel_launch(void* const* d_in, const int* in_sizes, int n_in,
                              void* d_out, int out_size, void* d_ws, size_t ws_size,
                              hipStream_t stream) {
    (void)n_in; (void)ws_size;
    const float* x    = (const float*)d_in[0];
    const int*   eidx = (const int*)  d_in[1];
    const float* ew   = (const float*)d_in[2];
    const int*   bat  = (const int*)  d_in[3];
    const float* W1   = (const float*)d_in[4];
    const float* b1   = (const float*)d_in[5];
    const float* W2   = (const float*)d_in[6];
    const float* b2   = (const float*)d_in[7];
    const float* W3   = (const float*)d_in[8];
    const float* b3   = (const float*)d_in[9];
    const float* linW = (const float*)d_in[10];
    const float* linb = (const float*)d_in[11];

    const int N = in_sizes[0] / FDIM;
    const int E = in_sizes[2];
    const int G = out_size / 10;
    const int* src = eidx;
    const int* dst = eidx + E;

    float* ws    = (float*)d_ws;
    float* dinv  = ws;  ws += N;              // deg, then dinv in place
    float* nrm   = ws;  ws += E;
    float* P     = ws;  ws += (size_t)N * HDIM;
    float* AGG   = ws;  ws += (size_t)N * HDIM;
    float* HB    = ws;  ws += (size_t)N * HDIM;
    float* POOL  = ws;  ws += (size_t)G * HDIM;
    _Float16* BP1 = (_Float16*)ws;            // 4096 halves (32B-aligned: all offsets /8)
    _Float16* BP2 = BP1 + 4 * 2 * 32 * 16;    // 1024 halves
    _Float16* BP3 = BP2 + 2 * 32 * 16;        // 1024 halves

    const int T = 256;
    auto blk = [](long long n, int t) { return (int)((n + t - 1) / t); };
    const int gemmBlocks = blk((long long)((N + 15) / 16) * 32, T);

    // weight fragment packing (tiny, once)
    k_pack_b<<<blk(4096, T), T, 0, stream>>>(W1, BP1, FDIM, 4);
    k_pack_b<<<blk(1024, T), T, 0, stream>>>(W2, BP2, HDIM, 1);
    k_pack_b<<<blk(1024, T), T, 0, stream>>>(W3, BP3, HDIM, 1);

    // degree normalization (once)
    k_zero_f32     <<<blk(N, T), T, 0, stream>>>(dinv, N);
    k_deg_edges    <<<blk(E, T), T, 0, stream>>>(dst, ew, dinv, E);
    k_finalize_dinv<<<blk(N, T), T, 0, stream>>>(dinv, N);
    k_edge_norm    <<<blk(E, T), T, 0, stream>>>(src, dst, ew, dinv, nrm, E);

    // layer 1
    k_gemm_f128 <<<gemmBlocks, T, 0, stream>>>(x, BP1, dinv, b1, P, AGG, N);
    k_edge_agg  <<<blk(E, T), T, 0, stream>>>(src, dst, nrm, P, AGG, E);
    k_norm_relu <<<blk(N, T), T, 0, stream>>>(AGG, HB, N);
    // layer 2
    k_gemm_f20  <<<gemmBlocks, T, 0, stream>>>(HB, BP2, dinv, b2, P, AGG, N);
    k_edge_agg  <<<blk(E, T), T, 0, stream>>>(src, dst, nrm, P, AGG, E);
    k_norm_relu <<<blk(N, T), T, 0, stream>>>(AGG, HB, N);
    // layer 3
    k_gemm_f20  <<<gemmBlocks, T, 0, stream>>>(HB, BP3, dinv, b3, P, AGG, N);
    k_edge_agg  <<<blk(E, T), T, 0, stream>>>(src, dst, nrm, P, AGG, E);
    k_norm_relu <<<blk(N, T), T, 0, stream>>>(AGG, HB, N);

    // pool + classifier
    k_zero_f32 <<<blk((long long)G * HDIM, T), T, 0, stream>>>(POOL, G * HDIM);
    k_pool_max <<<blk((long long)N * HDIM, T), T, 0, stream>>>(HB, bat, POOL, N);
    k_linear   <<<blk((long long)G * 10, T), T, 0, stream>>>(POOL, linW, linb, (float*)d_out, G);
}
